// BiEncoderModel_50242527428853
// MI455X (gfx1250) — matrix operand order repaired and checked by reading, not compiled, not run on hardware
//
#include <hip/hip_runtime.h>
#include <hip/hip_bf16.h>

typedef __attribute__((ext_vector_type(16))) __bf16 v16bf;
typedef __attribute__((ext_vector_type(8)))  float  v8f;

#define D    768
#define QI   64
#define PJ   256
#define NQ   16
#define NP   128
#define KC   64
#define NKC  (D / KC)    // 12
#define PAD  8           // row pitch 72 bf16 = 144 B (16B multiple, conflict-free)

// fp32 -> bf16, round-to-nearest-even, pure bit math
static __device__ __forceinline__ unsigned short f2bf(float f) {
  unsigned u = __builtin_bit_cast(unsigned, f);
  unsigned r = u + 0x7FFFu + ((u >> 16) & 1u);
  return (unsigned short)(r >> 16);
}
static __device__ __forceinline__ unsigned pack2(float a, float b) {
  return (unsigned)f2bf(a) | ((unsigned)f2bf(b) << 16);
}
static __device__ __forceinline__ float wave_max(float v) {
#pragma unroll
  for (int o = 16; o; o >>= 1) v = fmaxf(v, __shfl_xor(v, o, 32));
  return v;
}
static __device__ __forceinline__ float wave_sum(float v) {
#pragma unroll
  for (int o = 16; o; o >>= 1) v += __shfl_xor(v, o, 32);
  return v;
}

// ---------------------------------------------------------------------------
// Kernel 1: inverse L2 norms of every (q,i) and (p,j) row. One wave per row.
// ---------------------------------------------------------------------------
__global__ __launch_bounds__(256)
void norm_kernel(const float* __restrict__ q, const float* __restrict__ p,
                 float* __restrict__ qinv, float* __restrict__ pinv) {
  const int wave = threadIdx.x >> 5, lane = threadIdx.x & 31;
  const int row = blockIdx.x * 8 + wave;
  const float* src;
  float* dst;
  int idx;
  if (row < NQ * QI) {
    src = q + (size_t)row * D; dst = qinv; idx = row;
  } else {
    idx = row - NQ * QI;
    if (idx >= NP * PJ) return;
    src = p + (size_t)idx * D; dst = pinv;
  }
  float ss = 0.f;
  for (int k = lane; k < D; k += 32) { float x = src[k]; ss = fmaf(x, x, ss); }
  ss = wave_sum(ss);
  if (lane == 0) dst[idx] = 1.f / fmaxf(sqrtf(ss), 1e-12f);
}

// ---------------------------------------------------------------------------
// Kernel 2: fused (qn . pn^T) WMMA GEMM with max epilogue -> mv_scores[q,p],
// plus the dense score q_reps[q].p_reps[p]. One workgroup per (p-group,q-group).
// 4 waves; each wave owns 4 Mtiles x 4 Ntiles (16 accumulators) of 64x256.
// Fragments load as 2x ds_load_b128 each (contiguous 16B halves per layout).
// ---------------------------------------------------------------------------
__global__ __launch_bounds__(128)
void simgemm_kernel(const float* __restrict__ q, const float* __restrict__ p,
                    const float* __restrict__ qinv, const float* __restrict__ pinv,
                    float* __restrict__ scores, float* __restrict__ mv) {
  __shared__ __bf16 As[QI][KC + PAD];
  __shared__ __bf16 Bs[PJ][KC + PAD];
  __shared__ float  wred[4];

  const int pg = blockIdx.x, qg = blockIdx.y;
  const int tid = threadIdx.x, lane = tid & 31, wave = tid >> 5;

  // ---- dense score: q_reps[qg] . p_reps[pg]  (rows i=0 / j=0), TEMPERATURE==1
  {
    const float* qr = q + (size_t)(qg * QI) * D;
    const float* pr = p + (size_t)(pg * PJ) * D;
    float s = 0.f;
    for (int k = tid; k < D; k += 128) s = fmaf(qr[k], pr[k], s);
    s = wave_sum(s);
    if (lane == 0) wred[wave] = s;
    __syncthreads();
    if (tid == 0) {
      float t = 0.f;
#pragma unroll
      for (int w = 0; w < 4; ++w) t += wred[w];
      scores[qg * NP + pg] = t;
    }
    __syncthreads();
  }

  v8f acc[4][4];
  {
    v8f z = {0.f, 0.f, 0.f, 0.f, 0.f, 0.f, 0.f, 0.f};
#pragma unroll
    for (int mt = 0; mt < 4; ++mt)
#pragma unroll
      for (int nl = 0; nl < 4; ++nl) acc[mt][nl] = z;
  }
  const int ntile0 = wave * 4;  // this wave's 4 N tiles (16 tiles of 16 cols total)

  for (int kc = 0; kc < NKC; ++kc) {
    const int k0 = kc * KC;
    // ---- stage A chunk (64x64): 1024 float4 / 128 threads = 8 each
#pragma unroll
    for (int it = 0; it < 8; ++it) {
      const int e = (tid + it * 128) * 4;          // element in [0,4096)
      const int r = e >> 6, k = e & 63;
      const float4 v = *(const float4*)(q + (size_t)(qg * QI + r) * D + k0 + k);
      const float s = qinv[qg * QI + r];
      uint2 w;
      w.x = pack2(v.x * s, v.y * s);
      w.y = pack2(v.z * s, v.w * s);
      *(uint2*)&As[r][k] = w;                      // ds_store_b64 (8B aligned)
    }
    // ---- stage B chunk (256x64): 4096 float4 / 128 threads = 32 each
#pragma unroll
    for (int it = 0; it < 32; ++it) {
      const int e = (tid + it * 128) * 4;          // element in [0,16384)
      const int r = e >> 6, k = e & 63;
      const float4 v = *(const float4*)(p + (size_t)(pg * PJ + r) * D + k0 + k);
      const float s = pinv[pg * PJ + r];
      uint2 w;
      w.x = pack2(v.x * s, v.y * s);
      w.y = pack2(v.z * s, v.w * s);
      *(uint2*)&Bs[r][k] = w;
    }
    if (kc + 1 < NKC) {  // hint next B chunk toward caches (global_prefetch_b8)
      __builtin_prefetch(p + (size_t)(pg * PJ + 2 * tid) * D + k0 + KC, 0, 3);
      __builtin_prefetch(p + (size_t)(pg * PJ + 2 * tid + 1) * D + k0 + KC, 0, 3);
    }
    __syncthreads();

#pragma unroll
    for (int ks = 0; ks < 2; ++ks) {               // two 16x16x32 steps per chunk
      // A fragments: lane 0-15 row M=lane, K {kb..kb+7, kb+16..kb+23}; kb=ks*32(+8 hi lanes)
      v16bf afrag[4];
      const int mrow = lane & 15;
      const int kbA  = ((lane & 16) ? 8 : 0) + ks * 32;
#pragma unroll
      for (int mt = 0; mt < 4; ++mt) {
        uint4* ap = (uint4*)&afrag[mt];
        ap[0] = *(const uint4*)&As[mt * 16 + mrow][kbA];        // ds_load_b128
        ap[1] = *(const uint4*)&As[mt * 16 + mrow][kbA + 16];   // ds_load_b128
      }
      // B fragments: lane 0-15 col N=lane, K {kb..kb+7, kb+16..kb+23}; kb=ks*32(+8 hi lanes)
      v16bf bfrag[4];
      const int ncol = lane & 15;
      const int kbB  = ((lane & 16) ? 8 : 0) + ks * 32;
#pragma unroll
      for (int nl = 0; nl < 4; ++nl) {
        uint4* bp = (uint4*)&bfrag[nl];
        bp[0] = *(const uint4*)&Bs[(ntile0 + nl) * 16 + ncol][kbB];
        bp[1] = *(const uint4*)&Bs[(ntile0 + nl) * 16 + ncol][kbB + 16];
      }
#pragma unroll
      for (int mt = 0; mt < 4; ++mt)
#pragma unroll
        for (int nl = 0; nl < 4; ++nl)
          acc[mt][nl] = __builtin_amdgcn_wmma_f32_16x16x32_bf16(
              false, afrag[mt], false, bfrag[nl], (short)0, acc[mt][nl],
              false, false);
    }
    __syncthreads();
  }

  // ---- epilogue: max over the whole 64x256 tile -> mv_scores[qg,pg]
  float m = -3.402823466e38f;
#pragma unroll
  for (int mt = 0; mt < 4; ++mt)
#pragma unroll
    for (int nl = 0; nl < 4; ++nl)
#pragma unroll
      for (int e = 0; e < 8; ++e) m = fmaxf(m, acc[mt][nl][e]);
  m = wave_max(m);
  if (lane == 0) wred[wave] = m;
  __syncthreads();
  if (tid == 0) {
    float t = wred[0];
#pragma unroll
    for (int w = 1; w < 4; ++w) t = fmaxf(t, wred[w]);
    mv[qg * NP + pg] = t;
  }
}

// ---------------------------------------------------------------------------
// Kernel 3: CE + KLD reduction over the 16x128 score matrices -> scalar loss.
// 16 waves, one wave per row (4 columns per lane).
// Note: reference overwrites loss_mulvec1 with the KLD before forming L1.
// ---------------------------------------------------------------------------
__global__ __launch_bounds__(512)
void loss_kernel(const float* __restrict__ scores, const float* __restrict__ mv,
                 float* __restrict__ out) {
  __shared__ float red[16][4];
  const int wave = threadIdx.x >> 5, lane = threadIdx.x & 31;
  const int qrow = wave;

  float s[4], m[4], si[4];
#pragma unroll
  for (int c = 0; c < 4; ++c) {
    const int col = lane + 32 * c;
    s[c]  = scores[qrow * NP + col];
    m[c]  = mv[qrow * NP + col];
    si[c] = 1.0f * s[c] + 0.3f * m[c];          // W1*scores + W3*mv_scores
  }
  auto lse = [&](const float* x) -> float {
    float mx = fmaxf(fmaxf(x[0], x[1]), fmaxf(x[2], x[3]));
    mx = wave_max(mx);
    float e = 0.f;
#pragma unroll
    for (int c = 0; c < 4; ++c) e += expf(x[c] - mx);
    e = wave_sum(e);
    return mx + logf(e);
  };
  const float lse_s = lse(s), lse_m = lse(m), lse_i = lse(si);

  const int tcol = qrow * (NP / NQ);             // target = q * group_size
  float ce_s = 0.f, ce_i = 0.f, klds = 0.f, kldm = 0.f;
#pragma unroll
  for (int c = 0; c < 4; ++c) {
    const int col = lane + 32 * c;
    const float t  = si[c] - lse_i;              // t_logits
    const float et = expf(t);
    klds += et * (t - (s[c] - lse_s));
    kldm += et * (t - (m[c] - lse_m));
    if (col == tcol) { ce_s = lse_s - s[c]; ce_i = lse_i - si[c]; }
  }
  klds = wave_sum(klds);
  kldm = wave_sum(kldm);
  ce_s = wave_sum(ce_s);
  ce_i = wave_sum(ce_i);
  if (lane == 0) {
    red[qrow][0] = ce_s; red[qrow][1] = ce_i;
    red[qrow][2] = klds; red[qrow][3] = kldm;
  }
  __syncthreads();
  if (threadIdx.x == 0) {
    float CEs = 0.f, CEi = 0.f, Ks = 0.f, Km = 0.f;
#pragma unroll
    for (int r = 0; r < 16; ++r) {
      CEs += red[r][0]; CEi += red[r][1]; Ks += red[r][2]; Km += red[r][3];
    }
    CEs *= (1.f / NQ); CEi *= (1.f / NQ); Ks *= (1.f / NQ); Km *= (1.f / NQ);
    const float L1 = 0.3f * (CEs + Km + CEi);    // loss_mulvec1 == KLD_m here
    const float L2 = 0.2f * (Ks + Km);
    out[0] = 0.5f * (L1 + L2);
  }
}

extern "C" void kernel_launch(void* const* d_in, const int* in_sizes, int n_in,
                              void* d_out, int out_size, void* d_ws, size_t ws_size,
                              hipStream_t stream) {
  const float* q = (const float*)d_in[0];   // (16,64,768) fp32
  const float* p = (const float*)d_in[1];   // (128,256,768) fp32
  float* ws     = (float*)d_ws;             // ~152 KB used
  float* qinv   = ws;                       // 1024
  float* pinv   = qinv + NQ * QI;           // 32768
  float* scores = pinv + NP * PJ;           // 2048
  float* mv     = scores + NQ * NP;         // 2048

  const int nrows = NQ * QI + NP * PJ;      // 33792 rows, 8 per block
  norm_kernel<<<(nrows + 7) / 8, 256, 0, stream>>>(q, p, qinv, pinv);

  dim3 grid(NP, NQ);                        // 2048 workgroups
  simgemm_kernel<<<grid, 128, 0, stream>>>(q, p, qinv, pinv, scores, mv);

  loss_kernel<<<1, 512, 0, stream>>>(scores, mv, (float*)d_out);
}
